// WMam_31190052503710
// MI455X (gfx1250) — matrix operand order repaired and checked by reading, not compiled, not run on hardware
//
#include <hip/hip_runtime.h>

typedef __attribute__((ext_vector_type(16))) __bf16 v16bf;
typedef __attribute__((ext_vector_type(8)))  __bf16 v8bf;
typedef __attribute__((ext_vector_type(8)))  float  v8f;
typedef __attribute__((ext_vector_type(4)))  float  v4f;

#define NWIN    2048
#define LTOK    64
#define DMODEL  128
#define DINNER  256

// LDS row strides (elements); padded so 16-row tile loads hit distinct banks
#define SEQ_STRIDE 136   // 128 + 8 bf16 -> 272B rows (bank shift 4/row)
#define BIG_STRIDE 264   // 256 + 8 bf16 -> 528B rows (bank shift 4/row)
#define XP_STRIDE  56    // 48 + 8 f32
#define OUT_STRIDE 132   // 128 + 4 f32

// LDS byte offsets (Y aliases SEQ: SEQ dead before scan writes Y)
#define OFF_Y    0                        // bf16 [64][264] = 33792 (SEQ uses first 17408)
#define OFF_U    33792                    // bf16 [64][264]  (x -> u in place; f32 OUTBUF later)
#define OFF_SZ   67584                    // bf16 [64][264]  silu(z)
#define OFF_XP   101376                   // f32  [64][56]   = 14336
#define SMEM_BYTES 115712

// workspace byte offsets
#define WS_A     0          // f32  [256][16]  = 16384   A = -exp(A_log)
#define WS_WINT  16384      // bf16 [512][128] = 131072  W_in^T
#define WS_WXT   147456     // bf16 [48][256]  = 24576   W_xproj^T (zero-padded 40->48)
#define WS_WOT   172032     // bf16 [128][256] = 65536   W_out^T

static __device__ __forceinline__ __bf16 f2bf(float f) {
  unsigned u = __float_as_uint(f);
  u += 0x7FFFu + ((u >> 16) & 1u);                // round-to-nearest-even
  unsigned short s = (unsigned short)(u >> 16);
  return __builtin_bit_cast(__bf16, s);
}
static __device__ __forceinline__ float bf2f(__bf16 h) {
  unsigned short s = __builtin_bit_cast(unsigned short, h);
  return __uint_as_float(((unsigned)s) << 16);
}
static __device__ __forceinline__ float silu_f(float v) {
  return v / (1.f + __expf(-v));
}

// A fragment (16x32 bf16) from LDS, row-major rows with given stride.
// lanes 0-15: M=lane, K in {k0..k0+15}; lanes 16-31: M=lane-16, K in {k0+16..k0+31}.
static __device__ __forceinline__ v16bf ldsA(const __bf16* base, int stride,
                                             int rowBase, int kBase, int lane) {
  const __bf16* p = base + (rowBase + (lane & 15)) * stride + kBase + ((lane < 16) ? 0 : 16);
  v8bf lo = *(const v8bf*)p;
  v8bf hi = *(const v8bf*)(p + 8);
  return __builtin_shufflevector(lo, hi, 0,1,2,3,4,5,6,7,8,9,10,11,12,13,14,15);
}
// B fragment (32x16 bf16) from global W^T (N-major): lane = N col, contiguous K run.
static __device__ __forceinline__ v16bf ldB(const __bf16* wT, int K,
                                            int nBase, int kBase, int lane) {
  const __bf16* p = wT + (size_t)(nBase + (lane & 15)) * K + kBase + ((lane < 16) ? 0 : 16);
  return *(const v16bf*)p;
}

#define WMMA_BF16(A, B, C) \
  __builtin_amdgcn_wmma_f32_16x16x32_bf16(false, (A), false, (B), (short)0, (C), false, false)

// ---------------- weight prep: transpose + bf16 + A = -exp(A_log) ----------------
__global__ void __launch_bounds__(256)
wmamba_prep(const float* __restrict__ W_in, const float* __restrict__ W_xproj,
            const float* __restrict__ W_out, const float* __restrict__ A_log,
            char* __restrict__ ws) {
  int tid = blockIdx.x * 256 + threadIdx.x;
  float*  Aw   = (float*)(ws + WS_A);
  __bf16* WinT = (__bf16*)(ws + WS_WINT);
  __bf16* WxT  = (__bf16*)(ws + WS_WXT);
  __bf16* WoT  = (__bf16*)(ws + WS_WOT);
  if (tid < 65536) {                       // W_in (128,512) -> [512][128]
    int n = tid >> 7, k = tid & 127;
    WinT[n * 128 + k] = f2bf(W_in[k * 512 + n]);
  } else if (tid < 65536 + 12288) {        // W_xproj (256,40) -> [48][256], pad 0
    int t = tid - 65536; int n = t >> 8, k = t & 255;
    WxT[n * 256 + k] = f2bf((n < 40) ? W_xproj[k * 40 + n] : 0.f);
  } else if (tid < 65536 + 12288 + 32768) {// W_out (256,128) -> [128][256]
    int t = tid - (65536 + 12288); int n = t >> 8, k = t & 255;
    WoT[n * 256 + k] = f2bf(W_out[k * 128 + n]);
  } else if (tid < 65536 + 12288 + 32768 + 4096) {
    int t = tid - (65536 + 12288 + 32768);
    Aw[t] = -__expf(A_log[t]);
  }
}

// ---------------- fused per-window mamba ----------------
__global__ void __launch_bounds__(256)
wmamba_main(const float* __restrict__ x, const float* __restrict__ pos,
            const float* __restrict__ conv_w, const float* __restrict__ conv_b,
            const float* __restrict__ W_dt, const float* __restrict__ b_dt,
            const float* __restrict__ Dp, const char* __restrict__ ws,
            float* __restrict__ out) {
  extern __shared__ char smem[];
  __bf16* SEQ = (__bf16*)(smem + OFF_Y);   // aliases Y (SEQ only live through GEMM1)
  __bf16* Y   = (__bf16*)(smem + OFF_Y);
  __bf16* U   = (__bf16*)(smem + OFF_U);
  __bf16* SZ  = (__bf16*)(smem + OFF_SZ);
  float*  XP  = (float*)(smem + OFF_XP);
  float*  OUTBUF = (float*)(smem + OFF_U); // aliases U (U/SZ dead after scan)

  const float*  Aw   = (const float*)(ws + WS_A);
  const __bf16* WinT = (const __bf16*)(ws + WS_WINT);
  const __bf16* WxT  = (const __bf16*)(ws + WS_WXT);
  const __bf16* WoT  = (const __bf16*)(ws + WS_WOT);

  const int tid  = threadIdx.x;
  const int wave = tid >> 5;
  const int lane = tid & 31;

  const int wi = blockIdx.x;
  const int bb = wi >> 10;
  const int nh = (wi >> 5) & 31;
  const int nw = wi & 31;
  const int inBase = bb * (DMODEL * 256 * 256) + (nh * 8) * 256 + nw * 8;

  // ---- stage window (+pos) into LDS bf16: SEQ[l][c], l = hs*8+ws ----
  // x is read exactly once -> non-temporal loads keep L2 for the shared weights
  for (int i = tid; i < LTOK * DMODEL; i += 256) {
    int c = i >> 6, l = i & 63;
    float v = __builtin_nontemporal_load(&x[inBase + c * 65536 + (l >> 3) * 256 + (l & 7)])
            + pos[c * 64 + l];
    SEQ[l * SEQ_STRIDE + c] = f2bf(v);
  }
  __syncthreads();

  // ---- GEMM1: xz = seq @ W_in (64x128x512); wave w owns N-tiles [4w,4w+4) ----
  const int mrowHalf = (lane < 16) ? 0 : 8;
  const int nlane = lane & 15;
  #pragma unroll 1
  for (int ni = 0; ni < 4; ++ni) {
    int nig = wave * 4 + ni;
    v8f a0c = {}, a1c = {}, a2c = {}, a3c = {};
    #pragma unroll
    for (int ki = 0; ki < 4; ++ki) {
      v16bf bf = ldB(WinT, 128, nig * 16, ki * 32, lane);
      v16bf f0 = ldsA(SEQ, SEQ_STRIDE, 0,  ki * 32, lane);
      v16bf f1 = ldsA(SEQ, SEQ_STRIDE, 16, ki * 32, lane);
      v16bf f2 = ldsA(SEQ, SEQ_STRIDE, 32, ki * 32, lane);
      v16bf f3 = ldsA(SEQ, SEQ_STRIDE, 48, ki * 32, lane);
      a0c = WMMA_BF16(f0, bf, a0c);
      a1c = WMMA_BF16(f1, bf, a1c);
      a2c = WMMA_BF16(f2, bf, a2c);
      a3c = WMMA_BF16(f3, bf, a3c);
    }
    bool isX = (nig < 16);
    __bf16* dst = isX ? U : SZ;
    int col = (isX ? nig : nig - 16) * 16 + nlane;
    #pragma unroll
    for (int r = 0; r < 8; ++r) {
      float v0 = a0c[r], v1 = a1c[r], v2 = a2c[r], v3 = a3c[r];
      if (!isX) { v0 = silu_f(v0); v1 = silu_f(v1); v2 = silu_f(v2); v3 = silu_f(v3); }
      int m = mrowHalf + r;
      dst[(m)      * BIG_STRIDE + col] = f2bf(v0);
      dst[(m + 16) * BIG_STRIDE + col] = f2bf(v1);
      dst[(m + 32) * BIG_STRIDE + col] = f2bf(v2);
      dst[(m + 48) * BIG_STRIDE + col] = f2bf(v3);
    }
  }
  __syncthreads();

  // ---- depthwise causal conv (D_CONV=4) + SiLU, in-place on U; thread = channel ----
  {
    int d = tid;
    float w0 = conv_w[d * 4 + 0], w1 = conv_w[d * 4 + 1];
    float w2 = conv_w[d * 4 + 2], w3 = conv_w[d * 4 + 3];
    float cb = conv_b[d];
    float xm1 = 0.f, xm2 = 0.f, xm3 = 0.f;
    #pragma unroll 4
    for (int t = 0; t < LTOK; ++t) {
      float xt = bf2f(U[t * BIG_STRIDE + d]);
      float a = cb + w0 * xm3 + w1 * xm2 + w2 * xm1 + w3 * xt;
      U[t * BIG_STRIDE + d] = f2bf(silu_f(a));
      xm3 = xm2; xm2 = xm1; xm1 = xt;
    }
  }
  __syncthreads();

  // ---- GEMM2: xp = u @ W_xproj^T (64x256x48); 12 tiles over all 8 waves ----
  #pragma unroll 1
  for (int tix = wave; tix < 12; tix += 8) {
    int mi = tix & 3, ni = tix >> 2;
    v8f acc = {};
    #pragma unroll
    for (int ki = 0; ki < 8; ++ki) {
      v16bf af = ldsA(U, BIG_STRIDE, mi * 16, ki * 32, lane);
      v16bf bf = ldB(WxT, 256, ni * 16, ki * 32, lane);
      acc = WMMA_BF16(af, bf, acc);
    }
    int col = ni * 16 + nlane;
    int mrow = mi * 16 + mrowHalf;
    #pragma unroll
    for (int r = 0; r < 8; ++r) XP[(mrow + r) * XP_STRIDE + col] = acc[r];
  }
  __syncthreads();

  // ---- selective scan: thread = channel d; 16 states in registers ----
  {
    int d = tid;
    float Areg[16], Bm[16], Cm[16], h[16];
    #pragma unroll
    for (int i = 0; i < 4; ++i) {
      v4f a4 = *(const v4f*)(Aw + d * 16 + i * 4);
      #pragma unroll
      for (int j = 0; j < 4; ++j) Areg[i * 4 + j] = a4[j];
    }
    float wdt[8];
    #pragma unroll
    for (int r = 0; r < 8; ++r) wdt[r] = W_dt[r * 256 + d];
    float bdt = b_dt[d], dp = Dp[d];
    #pragma unroll
    for (int n = 0; n < 16; ++n) h[n] = 0.f;

    #pragma unroll 1
    for (int t = 0; t < LTOK; ++t) {
      const float* row = XP + t * XP_STRIDE;
      v4f q0 = *(const v4f*)(row + 0);
      v4f q1 = *(const v4f*)(row + 4);
      #pragma unroll
      for (int i = 0; i < 4; ++i) {
        v4f qb = *(const v4f*)(row + 8 + i * 4);
        v4f qc = *(const v4f*)(row + 24 + i * 4);
        #pragma unroll
        for (int j = 0; j < 4; ++j) { Bm[i * 4 + j] = qb[j]; Cm[i * 4 + j] = qc[j]; }
      }
      float dtp = bdt + q0[0] * wdt[0] + q0[1] * wdt[1] + q0[2] * wdt[2] + q0[3] * wdt[3]
                      + q1[0] * wdt[4] + q1[1] * wdt[5] + q1[2] * wdt[6] + q1[3] * wdt[7];
      float dt = (dtp > 15.f) ? dtp : __logf(1.f + __expf(dtp));   // softplus
      float ut = bf2f(U[t * BIG_STRIDE + d]);
      float du = dt * ut;
      float y = 0.f;
      #pragma unroll
      for (int n = 0; n < 16; ++n) {
        float dA = __expf(dt * Areg[n]);
        h[n] = dA * h[n] + du * Bm[n];
        y += h[n] * Cm[n];
      }
      y += ut * dp;
      y *= bf2f(SZ[t * BIG_STRIDE + d]);      // silu(z) already applied
      Y[t * BIG_STRIDE + d] = f2bf(y);
    }
  }
  __syncthreads();

  // ---- GEMM3: out = y @ W_out^T (64x256x128); wave w owns N-tile w ----
  {
    int dm = wave * 16 + nlane;
    const __bf16* wrow = WoT + (size_t)dm * 256;
    #pragma unroll 1
    for (int mi = 0; mi < 4; ++mi) {
      v8f acc = {};
      #pragma unroll
      for (int ki = 0; ki < 8; ++ki) {
        v16bf af = ldsA(Y, BIG_STRIDE, mi * 16, ki * 32, lane);
        v16bf bf = *(const v16bf*)(wrow + ki * 32 + ((lane < 16) ? 0 : 16));
        acc = WMMA_BF16(af, bf, acc);
      }
      int t0 = mi * 16 + mrowHalf;
      #pragma unroll
      for (int r = 0; r < 8; ++r) OUTBUF[(t0 + r) * OUT_STRIDE + dm] = acc[r];
    }
  }
  __syncthreads();

  // ---- staged write-out: 32B-contiguous segments along w, non-temporal ----
  {
    const int outBase = bb * 8388608 + nh * 2048 + nw * 8;
    for (int i = tid; i < LTOK * DMODEL; i += 256) {
      int dm = i >> 6, l = i & 63;
      __builtin_nontemporal_store(OUTBUF[l * OUT_STRIDE + dm],
                                  &out[outBase + dm * 65536 + (l >> 3) * 256 + (l & 7)]);
    }
  }
}

extern "C" void kernel_launch(void* const* d_in, const int* in_sizes, int n_in,
                              void* d_out, int out_size, void* d_ws, size_t ws_size,
                              hipStream_t stream) {
  (void)in_sizes; (void)n_in; (void)out_size; (void)ws_size;
  const float* x       = (const float*)d_in[0];
  const float* pos     = (const float*)d_in[1];
  const float* W_in    = (const float*)d_in[2];
  const float* conv_w  = (const float*)d_in[3];
  const float* conv_b  = (const float*)d_in[4];
  const float* W_xproj = (const float*)d_in[5];
  const float* W_dt    = (const float*)d_in[6];
  const float* b_dt    = (const float*)d_in[7];
  const float* A_log   = (const float*)d_in[8];
  const float* Dp      = (const float*)d_in[9];
  const float* W_out   = (const float*)d_in[10];
  float* out = (float*)d_out;
  char*  ws  = (char*)d_ws;

  // weight prep: 65536 + 12288 + 32768 + 4096 = 114688 threads
  wmamba_prep<<<448, 256, 0, stream>>>(W_in, W_xproj, W_out, A_log, ws);

  hipFuncSetAttribute(reinterpret_cast<const void*>(wmamba_main),
                      hipFuncAttributeMaxDynamicSharedMemorySize, SMEM_BYTES);
  wmamba_main<<<NWIN, 256, SMEM_BYTES, stream>>>(x, pos, conv_w, conv_b,
                                                 W_dt, b_dt, Dp, ws, out);
}
